// GraphSAGE_39058432590076
// MI455X (gfx1250) — matrix operand order repaired and checked by reading, not compiled, not run on hardware
//
#include <hip/hip_runtime.h>

typedef _Float16 v16h __attribute__((ext_vector_type(16)));
typedef _Float16 v8h  __attribute__((ext_vector_type(8)));
typedef float    v8f  __attribute__((ext_vector_type(8)));

#define NNODES 50000
#define KNEI   16
#define FEAT   128

// ---------------------------------------------------------------------------
// Pack W[256, Fout] (f32, row-major) into f16 WMMA B-fragments:
//   dst[((c*8 + s)*32 + lane)*16 + e] = (f16) W[(s*32 + 8*(lane>>4) + e + 8*(e>>3))*Fout + c*16 + (lane&15)]
// so each lane of a wave reads its 16 B-halfs as one contiguous 32B chunk.
// ---------------------------------------------------------------------------
__global__ void pack_w_kernel(const float* __restrict__ W, _Float16* __restrict__ dst,
                              int Fout, int total) {
    int p = blockIdx.x * blockDim.x + threadIdx.x;
    if (p >= total) return;
    int e = p & 15;
    int l = (p >> 4) & 31;
    int s = (p >> 9) & 7;
    int c = p >> 12;
    int Kr  = s * 32 + ((l >> 4) << 3) + e + ((e >> 3) << 3);
    int col = (c << 4) + (l & 15);
    dst[p] = (_Float16)W[Kr * Fout + col];
}

// ---------------------------------------------------------------------------
// One GraphSAGE layer: 16 nodes / block, 128 threads (4 waves).
// Phase 1: gather-mean + concat into LDS f16 tile [16 x 256] (padded rows).
// Phase 2: WMMA f32_16x16x32_f16, K = 256 in 8 steps; FOUT/16 col-tiles
//          distributed over the 4 waves.
// ---------------------------------------------------------------------------
template <bool IN_F16, int FOUT, bool RELU, bool OUT_F16>
__global__ __launch_bounds__(128)
void sage_layer_kernel(const void* __restrict__ hin_v,
                       const int* __restrict__ adj,
                       const _Float16* __restrict__ wpack,
                       void* __restrict__ hout_v) {
    constexpr int ROW = 264;  // 256 + 8 halfs pad -> 528B row stride, spreads LDS banks
    __shared__ __align__(16) _Float16 sh[16 * ROW];

    const int t     = threadIdx.x;
    const int nbase = blockIdx.x * 16;

    // ---------------- Phase 1: self + mean(neighbors) -> LDS f16 -----------
    {
        const int m    = t >> 3;          // row 0..15
        const int f0   = (t & 7) << 4;    // 16-feature chunk
        const int node = nbase + m;

        float self[16];
        float acc[16];
#pragma unroll
        for (int j = 0; j < 16; ++j) acc[j] = 0.0f;

        if constexpr (IN_F16) {
            const _Float16* hin = (const _Float16*)hin_v;
            const v8h* sp = (const v8h*)(hin + (size_t)node * FEAT + f0);
            v8h s0 = sp[0], s1 = sp[1];
#pragma unroll
            for (int j = 0; j < 8; ++j) { self[j] = (float)s0[j]; self[j + 8] = (float)s1[j]; }
#pragma unroll
            for (int k = 0; k < KNEI; ++k) {
                const int row = adj[k * NNODES + node];
                const v8h* rp = (const v8h*)(hin + (size_t)row * FEAT + f0);
                v8h a0 = rp[0], a1 = rp[1];
#pragma unroll
                for (int j = 0; j < 8; ++j) { acc[j] += (float)a0[j]; acc[j + 8] += (float)a1[j]; }
            }
        } else {
            const float* hin = (const float*)hin_v;
            const float4* sp = (const float4*)(hin + (size_t)node * FEAT + f0);
#pragma unroll
            for (int q = 0; q < 4; ++q) {
                float4 s = sp[q];
                self[q * 4 + 0] = s.x; self[q * 4 + 1] = s.y;
                self[q * 4 + 2] = s.z; self[q * 4 + 3] = s.w;
            }
#pragma unroll
            for (int k = 0; k < KNEI; ++k) {
                const int row = adj[k * NNODES + node];
                const float4* rp = (const float4*)(hin + (size_t)row * FEAT + f0);
#pragma unroll
                for (int q = 0; q < 4; ++q) {
                    float4 a = rp[q];
                    acc[q * 4 + 0] += a.x; acc[q * 4 + 1] += a.y;
                    acc[q * 4 + 2] += a.z; acc[q * 4 + 3] += a.w;
                }
            }
        }
#pragma unroll
        for (int j = 0; j < 16; ++j) {
            sh[m * ROW + f0 + j]        = (_Float16)self[j];
            sh[m * ROW + FEAT + f0 + j] = (_Float16)(acc[j] * (1.0f / (float)KNEI));
        }
    }
    __syncthreads();

    // ---------------- Phase 2: WMMA GEMM hcat[16x256] @ W[256xFOUT] --------
    const int lane = t & 31;
    const int wave = t >> 5;
    const int m16  = lane & 15;
    const int half = lane >> 4;
    constexpr int CT = FOUT / 16;

    for (int c = wave; c < CT; c += 4) {
        v8f acc = {};
#pragma unroll
        for (int s = 0; s < 8; ++s) {
            // A fragment (16x32 f16): lane row = m16; e<8 -> K=8*half+e, e>=8 -> K=16+8*half+(e-8)
            const _Float16* ap = &sh[m16 * ROW + s * 32 + 8 * half];
            v8h lo = *(const v8h*)ap;
            v8h hi = *(const v8h*)(ap + 16);
            v16h a;
#pragma unroll
            for (int j = 0; j < 8; ++j) { a[j] = lo[j]; a[j + 8] = hi[j]; }
            // B fragment: pre-packed, 32B contiguous per lane
            v16h b = *(const v16h*)(wpack + ((((c * 8 + s) * 32) + lane) << 4));
            acc = __builtin_amdgcn_wmma_f32_16x16x32_f16(
                false, a, false, b, (short)0, acc, false, false);
        }
        // D layout: element r -> (M = r + 8*half, N = m16)
#pragma unroll
        for (int r = 0; r < 8; ++r) {
            const int M   = r + 8 * half;
            const int col = c * 16 + m16;
            float v = acc[r];
            if constexpr (RELU) v = fmaxf(v, 0.0f);
            if constexpr (OUT_F16) {
                ((_Float16*)hout_v)[(size_t)(nbase + M) * FOUT + col] = (_Float16)v;
            } else {
                ((float*)hout_v)[(size_t)(nbase + M) * FOUT + col] = v;
            }
        }
    }
}

// ---------------------------------------------------------------------------
// Workspace layout (bytes):
//   [0,       163840)   packed f16 weights: W0p(64KB) W1p(64KB) W2p(32KB)
//   [163840, +12.8MB)   h0 f16 [N x 128]
//   [ ...  , +12.8MB)   h1 f16 [N x 128]
// Total ~24.6 MiB.
// ---------------------------------------------------------------------------
extern "C" void kernel_launch(void* const* d_in, const int* in_sizes, int n_in,
                              void* d_out, int out_size, void* d_ws, size_t ws_size,
                              hipStream_t stream) {
    (void)in_sizes; (void)n_in; (void)out_size; (void)ws_size;

    const float* x   = (const float*)d_in[0];
    const int*   adj = (const int*)d_in[1];
    const float* W0  = (const float*)d_in[2];
    const float* W1  = (const float*)d_in[3];
    const float* W2  = (const float*)d_in[4];
    float*       out = (float*)d_out;

    char* ws = (char*)d_ws;
    _Float16* w0p = (_Float16*)ws;              // 32768 halfs
    _Float16* w1p = w0p + 32768;                // 32768 halfs
    _Float16* w2p = w1p + 32768;                // 16384 halfs
    _Float16* h0  = (_Float16*)(ws + 163840);
    _Float16* h1  = h0 + (size_t)NNODES * FEAT;

    pack_w_kernel<<<(32768 + 255) / 256, 256, 0, stream>>>(W0, w0p, 128, 32768);
    pack_w_kernel<<<(32768 + 255) / 256, 256, 0, stream>>>(W1, w1p, 128, 32768);
    pack_w_kernel<<<(16384 + 255) / 256, 256, 0, stream>>>(W2, w2p, 64, 16384);

    const int blocks = NNODES / 16;  // 3125 (exact)
    sage_layer_kernel<false, 128, true,  true ><<<blocks, 128, 0, stream>>>(x,  adj, w0p, h0);
    sage_layer_kernel<true,  128, true,  true ><<<blocks, 128, 0, stream>>>(h0, adj, w1p, h1);
    sage_layer_kernel<true,   64, false, false><<<blocks, 128, 0, stream>>>(h1, adj, w2p, out);
}
